// Decoder_9474697855725
// MI455X (gfx1250) — matrix operand order repaired and checked by reading, not compiled, not run on hardware
//
#include <hip/hip_runtime.h>
#include <math.h>

typedef _Float16 v16h __attribute__((ext_vector_type(16)));
typedef _Float16 v8h  __attribute__((ext_vector_type(8)));
typedef float    v8f  __attribute__((ext_vector_type(8)));

constexpr int NB = 64;
constexpr int NS = 2048;
constexpr int NE = 256;
constexpr int NR = 50;
constexpr int OUTW = NR + 4 * NS;   // 8242

__device__ __forceinline__ float wave_sum(float v) {
#pragma unroll
  for (int off = 16; off > 0; off >>= 1) v += __shfl_xor(v, off, 32);
  return v;
}
__device__ __forceinline__ float sigmoidf(float x) { return 1.f / (1.f + __expf(-x)); }

// ---------------------------------------------------------------- Kt prep
// Kt[n][k] (f16), k = w*256 + i, = W_conv[n, i, w]  (encoder-channel taps,
// transposed so B-fragment loads are contiguous 32B per lane)
__global__ __launch_bounds__(256) void prep_kt_kernel(const float* __restrict__ W_conv,
                                                      _Float16* __restrict__ Kt) {
  int id = blockIdx.x * 256 + threadIdx.x;   // 768 blocks -> 196608 = 256*768
  int n = id / 768;
  int k = id - n * 768;
  int w = k >> 8;
  int i = k & 255;
  Kt[id] = (_Float16)W_conv[((size_t)n * (2 * NE) + i) * 3 + w];
}

// ---------------------------------------------------------------- U = conv(enc part)
// One workgroup = 16 output rows (one b, 16 consecutive s) x all 256 cols.
// 8 waves, 2 N-tiles each, K-loop 768 in steps of 32 -> 48 WMMAs / wave.
__global__ __launch_bounds__(256) void conv_u_wmma(const float* __restrict__ enc,
                                                   const _Float16* __restrict__ Kt,
                                                   float* __restrict__ U) {
  __shared__ __align__(16) _Float16 Alds[16 * 768];
  const int wg = blockIdx.x;          // 8192 = 64 * 128
  const int b  = wg >> 7;
  const int s0 = (wg & 127) << 4;
  const int tid = threadIdx.x;

  // im2col fill: row m, k = w*256+i  -> enc[b, s0+m+w-1, i] (zero padded)
  for (int idx = tid; idx < 16 * 768; idx += 256) {
    int m = idx / 768;
    int k = idx - m * 768;
    int w = k >> 8;
    int i = k & 255;
    int srow = s0 + m + w - 1;
    float v = 0.f;
    if (srow >= 0 && srow < NS) v = enc[((size_t)b * NS + srow) * NE + i];
    Alds[idx] = (_Float16)v;
  }
  __syncthreads();

  const int lane = tid & 31;
  const int wave = tid >> 5;
  const int mn   = lane & 15;   // M row (A/D) and N col (B/D)
  const int half = lane >> 4;   // K-half select (A/B), M-half select (D)
  const int nt0  = wave * 2;

  const _Float16* kt0  = Kt + (size_t)(nt0 * 16 + mn) * 768 + half * 16;
  const _Float16* kt1  = kt0 + (size_t)16 * 768;
  const _Float16* a_lo = Alds + mn * 768 + half * 16;
  const _Float16* a_hi = a_lo + 8;

  v8f acc0 = {}; v8f acc1 = {};
#pragma unroll
  for (int kk = 0; kk < 24; ++kk) {
    const int kb = kk * 32;
    if (kk + 4 < 24) __builtin_prefetch(kt0 + kb + 128, 0, 1);  // global_prefetch_b8
    v8h lo = *(const v8h*)(a_lo + kb);
    v8h hi = *(const v8h*)(a_hi + kb);
    v16h a;
#pragma unroll
    for (int j = 0; j < 8; ++j) { a[j] = lo[j]; a[8 + j] = hi[j]; }
    v16h b0 = *(const v16h*)(kt0 + kb);
    v16h b1 = *(const v16h*)(kt1 + kb);
    acc0 = __builtin_amdgcn_wmma_f32_16x16x32_f16(false, a, false, b0, (short)0, acc0, false, false);
    acc1 = __builtin_amdgcn_wmma_f32_16x16x32_f16(false, a, false, b1, (short)0, acc1, false, false);
  }

  float* ubase = U + ((size_t)b * NS + s0) * NE;
#pragma unroll
  for (int r = 0; r < 8; ++r) {
    int m = r + 8 * half;
    ubase[(size_t)m * NE + nt0 * 16 + mn]       = acc0[r];
    ubase[(size_t)m * NE + (nt0 + 1) * 16 + mn] = acc1[r];
  }
}

// ---------------------------------------------------------------- tiny chain kernels
__global__ void init_hc_kernel(const float* __restrict__ h0, const float* __restrict__ c0,
                               float* __restrict__ h, float* __restrict__ c) {
  int g = blockIdx.x * 256 + threadIdx.x;   // NB*NE
  h[g] = h0[g];
  c[g] = c0[g];
}

__global__ void build_x1_kernel(const float* __restrict__ sos, float* __restrict__ x) {
  int g = blockIdx.x * 256 + threadIdx.x;
  x[g] = sos[g & (NE - 1)];
}

__global__ void build_x2_kernel(const float* __restrict__ rel, const int* __restrict__ r_in,
                                float* __restrict__ x) {
  int g = blockIdx.x * 256 + threadIdx.x;
  int b = g >> 8, e = g & 255;
  x[g] = rel[(size_t)r_in[b] * NE + e];
}

__global__ void build_x3_kernel(const float* __restrict__ enc, const int* __restrict__ k1,
                                const int* __restrict__ k2, float* __restrict__ x) {
  int g = blockIdx.x * 256 + threadIdx.x;
  int b = g >> 8, e = g & 255;
  x[g] = enc[((size_t)b * NS + k1[b]) * NE + e] + enc[((size_t)b * NS + k2[b]) * NE + e];
}

__global__ void lstm_kernel(const float* __restrict__ x, const float* __restrict__ h,
                            const float* __restrict__ c,
                            const float* __restrict__ W_ih, const float* __restrict__ W_hh,
                            const float* __restrict__ b_ih, const float* __restrict__ b_hh,
                            float* __restrict__ hn, float* __restrict__ cn) {
  int g = blockIdx.x * 256 + threadIdx.x;   // NB*NE
  int b = g >> 8, e = g & 255;
  const float* xb = x + (size_t)b * NE;
  const float* hb = h + (size_t)b * NE;
  const float* wi0 = W_ih + (size_t)e * NE;
  const float* wi1 = W_ih + (size_t)(NE + e) * NE;
  const float* wi2 = W_ih + (size_t)(2 * NE + e) * NE;
  const float* wi3 = W_ih + (size_t)(3 * NE + e) * NE;
  const float* wh0 = W_hh + (size_t)e * NE;
  const float* wh1 = W_hh + (size_t)(NE + e) * NE;
  const float* wh2 = W_hh + (size_t)(2 * NE + e) * NE;
  const float* wh3 = W_hh + (size_t)(3 * NE + e) * NE;
  float gi = 0.f, gf = 0.f, gg = 0.f, go = 0.f;
  for (int j = 0; j < NE; ++j) {
    float xv = xb[j], hv = hb[j];
    gi += wi0[j] * xv + wh0[j] * hv;
    gf += wi1[j] * xv + wh1[j] * hv;
    gg += wi2[j] * xv + wh2[j] * hv;
    go += wi3[j] * xv + wh3[j] * hv;
  }
  gi += b_ih[e] + b_hh[e];
  gf += b_ih[NE + e] + b_hh[NE + e];
  gg += b_ih[2 * NE + e] + b_hh[2 * NE + e];
  go += b_ih[3 * NE + e] + b_hh[3 * NE + e];
  float cc = sigmoidf(gf) * c[g] + sigmoidf(gi) * tanhf(gg);
  cn[g] = cc;
  hn[g] = sigmoidf(go) * tanhf(cc);
}

// one wave per (b,s): scores[b,s] = q[b] . enc[b,s]
__global__ __launch_bounds__(256) void scores_kernel(const float* __restrict__ enc,
                                                     const float* __restrict__ q,
                                                     float* __restrict__ sc) {
  int gw = blockIdx.x * 8 + (threadIdx.x >> 5);   // NB*NS waves
  int lane = threadIdx.x & 31;
  int b = gw >> 11, s = gw & (NS - 1);
  const float* er = enc + ((size_t)b * NS + s) * NE;
  const float* qb = q + (size_t)b * NE;
  float acc = 0.f;
#pragma unroll
  for (int j = 0; j < 8; ++j) {
    int e = j * 32 + lane;
    acc += qb[e] * er[e];
  }
  acc = wave_sum(acc);
  if (lane == 0) sc[gw] = acc;
}

__global__ __launch_bounds__(256) void softmax_kernel(float* __restrict__ sc) {
  __shared__ float red[256];
  int b = blockIdx.x, t = threadIdx.x;
  float* row = sc + (size_t)b * NS;
  float m = -1e30f;
#pragma unroll
  for (int j = 0; j < 8; ++j) m = fmaxf(m, row[j * 256 + t]);
  red[t] = m; __syncthreads();
  for (int off = 128; off > 0; off >>= 1) {
    if (t < off) red[t] = fmaxf(red[t], red[t + off]);
    __syncthreads();
  }
  m = red[0]; __syncthreads();
  float sum = 0.f;
#pragma unroll
  for (int j = 0; j < 8; ++j) {
    float ev = __expf(row[j * 256 + t] - m);
    row[j * 256 + t] = ev;
    sum += ev;
  }
  red[t] = sum; __syncthreads();
  for (int off = 128; off > 0; off >>= 1) {
    if (t < off) red[t] += red[t + off];
    __syncthreads();
  }
  float inv = 1.f / red[0];
#pragma unroll
  for (int j = 0; j < 8; ++j) row[j * 256 + t] *= inv;
}

__global__ void mix_kernel(const float* __restrict__ enc, const float* __restrict__ attn,
                           float* __restrict__ mix) {
  int g = blockIdx.x * 256 + threadIdx.x;   // NB*NE
  int b = g >> 8, e = g & 255;
  const float* a = attn + (size_t)b * NS;
  const float* er = enc + (size_t)b * NS * NE + e;
  float acc = 0.f;
  for (int s = 0; s < NS; ++s) acc += a[s] * er[(size_t)s * NE];
  mix[g] = acc;
}

__global__ void combine_kernel(const float* __restrict__ mix, const float* __restrict__ q,
                               const float* __restrict__ W_attn, const float* __restrict__ b_attn,
                               float* __restrict__ o) {
  int g = blockIdx.x * 256 + threadIdx.x;
  int b = g >> 8, e = g & 255;
  const float* wr = W_attn + (size_t)e * (2 * NE);
  const float* mb = mix + (size_t)b * NE;
  const float* qb = q + (size_t)b * NE;
  float acc = b_attn[e];
  for (int j = 0; j < NE; ++j) acc += wr[j] * mb[j];
  for (int j = 0; j < NE; ++j) acc += wr[NE + j] * qb[j];
  o[g] = tanhf(acc);
}

__global__ void t1_kernel(const float* __restrict__ o1, const float* __restrict__ W_rel,
                          const float* __restrict__ b_rel, float* __restrict__ out) {
  int g = blockIdx.x * 256 + threadIdx.x;
  if (g >= NB * NR) return;
  int b = g / NR, r = g - b * NR;
  const float* ob = o1 + (size_t)b * NE;
  const float* wr = W_rel + (size_t)r * NE;
  float acc = b_rel[r];
  for (int j = 0; j < NE; ++j) acc += wr[j] * ob[j];
  out[(size_t)b * OUTW + r] = acc;
}

// v[b,n] = o[b] @ Ko(tap-sum variants) + b_conv, for full/left/right edges
__global__ void vheads_kernel(const float* __restrict__ o, const float* __restrict__ W_conv,
                              const float* __restrict__ b_conv, float* __restrict__ vf,
                              float* __restrict__ vl, float* __restrict__ vr) {
  int g = blockIdx.x * 256 + threadIdx.x;   // NB*NE
  int b = g >> 8, n = g & 255;
  const float* ob = o + (size_t)b * NE;
  const float* base = W_conv + ((size_t)n * (2 * NE) + NE) * 3;
  float a0 = 0.f, a1 = 0.f, a2 = 0.f;
  for (int i = 0; i < NE; ++i) {
    float ov = ob[i];
    const float* w3 = base + (size_t)i * 3;
    a0 += ov * w3[0];
    a1 += ov * w3[1];
    a2 += ov * w3[2];
  }
  float bc = b_conv[n];
  vf[g] = a0 + a1 + a2 + bc;   // interior
  vl[g] = a1 + a2 + bc;        // s = 0
  vr[g] = a0 + a1 + bc;        // s = S-1
}

// one wave per (b,s): both ent_heads calls in a single pass over U
__global__ __launch_bounds__(256) void ent_heads_kernel(
    const float* __restrict__ U,
    const float* __restrict__ v2f, const float* __restrict__ v2l, const float* __restrict__ v2r,
    const float* __restrict__ v3f, const float* __restrict__ v3l, const float* __restrict__ v3r,
    const float* __restrict__ W1, const float* __restrict__ b1,
    const float* __restrict__ W2, const float* __restrict__ b2,
    float* __restrict__ out) {
  int gw = blockIdx.x * 8 + (threadIdx.x >> 5);   // NB*NS waves
  int lane = threadIdx.x & 31;
  int b = gw >> 11, s = gw & (NS - 1);
  const float* v2 = ((s == 0) ? v2l : (s == NS - 1) ? v2r : v2f) + (size_t)b * NE;
  const float* v3 = ((s == 0) ? v3l : (s == NS - 1) ? v3r : v3f) + (size_t)b * NE;
  const float* ur = U + ((size_t)b * NS + s) * NE;
  float a1a = 0.f, a2a = 0.f, a1b = 0.f, a2b = 0.f;
#pragma unroll
  for (int j = 0; j < 8; ++j) {
    int e = j * 32 + lane;
    float u = ur[e];
    float w1 = W1[e], w2 = W2[e];
    float r2 = fmaxf(u + v2[e], 0.f);
    float r3 = fmaxf(u + v3[e], 0.f);
    a1a += r2 * w1;  a2a += r2 * w2;
    a1b += r3 * w1;  a2b += r3 * w2;
  }
  a1a = wave_sum(a1a); a2a = wave_sum(a2a);
  a1b = wave_sum(a1b); a2b = wave_sum(a2b);
  if (lane == 0) {
    float* ob = out + (size_t)b * OUTW + NR;
    ob[s]          = a1a + b1[0];
    ob[NS + s]     = a2a + b2[0];
    ob[2 * NS + s] = a1b + b1[0];
    ob[3 * NS + s] = a2b + b2[0];
  }
}

extern "C" void kernel_launch(void* const* d_in, const int* in_sizes, int n_in,
                              void* d_out, int out_size, void* d_ws, size_t ws_size,
                              hipStream_t stream) {
  (void)in_sizes; (void)n_in; (void)out_size; (void)ws_size;
  const float* enc    = (const float*)d_in[0];
  const float* h0     = (const float*)d_in[1];
  const float* c0     = (const float*)d_in[2];
  const int*   r_in   = (const int*)d_in[3];
  const int*   k1     = (const int*)d_in[4];
  const int*   k2     = (const int*)d_in[5];
  const float* W_ih   = (const float*)d_in[6];
  const float* W_hh   = (const float*)d_in[7];
  const float* b_ih   = (const float*)d_in[8];
  const float* b_hh   = (const float*)d_in[9];
  const float* W_attn = (const float*)d_in[10];
  const float* b_attn = (const float*)d_in[11];
  const float* W_conv = (const float*)d_in[12];
  const float* b_conv = (const float*)d_in[13];
  const float* sos    = (const float*)d_in[14];
  const float* rel    = (const float*)d_in[15];
  const float* W_rel  = (const float*)d_in[16];
  const float* b_rel  = (const float*)d_in[17];
  const float* W_ent1 = (const float*)d_in[18];
  const float* b_ent1 = (const float*)d_in[19];
  const float* W_ent2 = (const float*)d_in[20];
  const float* b_ent2 = (const float*)d_in[21];
  float* out = (float*)d_out;

  char* p = (char*)d_ws;
  auto alloc = [&](size_t bytes) -> void* {
    void* r = (void*)p;
    p += (bytes + 255) & ~(size_t)255;
    return r;
  };
  _Float16* Kt = (_Float16*)alloc((size_t)256 * 768 * 2);
  float* U     = (float*)alloc((size_t)NB * NS * NE * 4);
  float* sc    = (float*)alloc((size_t)NB * NS * 4);
  const size_t BE = (size_t)NB * NE * 4;
  float* x    = (float*)alloc(BE);
  float* hP   = (float*)alloc(BE);
  float* cP   = (float*)alloc(BE);
  float* hQ   = (float*)alloc(BE);
  float* cQ   = (float*)alloc(BE);
  float* mixb = (float*)alloc(BE);
  float* oatt = (float*)alloc(BE);
  float* v2f  = (float*)alloc(BE);
  float* v2l  = (float*)alloc(BE);
  float* v2r  = (float*)alloc(BE);
  float* v3f  = (float*)alloc(BE);
  float* v3l  = (float*)alloc(BE);
  float* v3r  = (float*)alloc(BE);

  const int beBlocks = NB * NE / 256;          // 64
  const int waveBlocks = NB * NS / 8;          // 16384

  // heavy path: U = conv(enc-part) via WMMA (independent of LSTM chain)
  prep_kt_kernel<<<768, 256, 0, stream>>>(W_conv, Kt);
  conv_u_wmma<<<NB * (NS / 16), 256, 0, stream>>>(enc, Kt, U);

  init_hc_kernel<<<beBlocks, 256, 0, stream>>>(h0, c0, hP, cP);

  float *h = hP, *c = cP, *hn = hQ, *cn = cQ;
  auto lstm_step = [&](void) {
    lstm_kernel<<<beBlocks, 256, 0, stream>>>(x, h, c, W_ih, W_hh, b_ih, b_hh, hn, cn);
    float* t;
    t = h; h = hn; hn = t;
    t = c; c = cn; cn = t;
  };
  auto attend = [&](void) {   // q = h (post-swap) -> oatt
    scores_kernel<<<waveBlocks, 256, 0, stream>>>(enc, h, sc);
    softmax_kernel<<<NB, 256, 0, stream>>>(sc);
    mix_kernel<<<beBlocks, 256, 0, stream>>>(enc, sc, mixb);
    combine_kernel<<<beBlocks, 256, 0, stream>>>(mixb, h, W_attn, b_attn, oatt);
  };

  // t1: <sos>
  build_x1_kernel<<<beBlocks, 256, 0, stream>>>(sos, x);
  lstm_step();
  attend();
  t1_kernel<<<(NB * NR + 255) / 256, 256, 0, stream>>>(oatt, W_rel, b_rel, out);

  // t2: rel_emb[r_in]
  build_x2_kernel<<<beBlocks, 256, 0, stream>>>(rel, r_in, x);
  lstm_step();
  attend();
  vheads_kernel<<<beBlocks, 256, 0, stream>>>(oatt, W_conv, b_conv, v2f, v2l, v2r);

  // t3: enc[k1]+enc[k2]
  build_x3_kernel<<<beBlocks, 256, 0, stream>>>(enc, k1, k2, x);
  lstm_step();
  attend();
  vheads_kernel<<<beBlocks, 256, 0, stream>>>(oatt, W_conv, b_conv, v3f, v3l, v3r);

  // both ent_heads calls in one pass over U
  ent_heads_kernel<<<waveBlocks, 256, 0, stream>>>(U, v2f, v2l, v2r, v3f, v3l, v3r,
                                                   W_ent1, b_ent1, W_ent2, b_ent2, out);
}
